// MixtureOfExperts_72035191488929
// MI455X (gfx1250) — matrix operand order repaired and checked by reading, not compiled, not run on hardware
//
#include <hip/hip_runtime.h>
#include <hip/hip_bf16.h>

// MoE: B=4096 tokens, D=3072, H=4096, E=8, top-2 routing.
// Grouped per-expert GEMMs with V_WMMA_F32_16X16X32_BF16 (wave32),
// bf16 LDS tiles (converted once at staging), double-buffered pipeline,
// async global->LDS copies for the bf16 activation tiles in FFN2.

#define B_TOK 4096
#define DMODEL 3072
#define HDIM 4096
#define NEXP 8
#define NSLOTS (B_TOK * 2)

#define MT 128
#define NT 128
#define KT 32
#define LDT 40  // bf16 LDS tile row stride (32 + 8 pad)

#define AS1 __attribute__((address_space(1)))
#define AS3 __attribute__((address_space(3)))

typedef int vi4 __attribute__((vector_size(16)));  // v4i32 (GCC-style, matches builtin)

#if __has_builtin(__builtin_amdgcn_global_load_async_to_lds_b128)
#define HAVE_ASYNC_LDS 1
#else
#define HAVE_ASYNC_LDS 0
#endif

#if __has_builtin(__builtin_amdgcn_s_wait_asynccnt)
#define WAIT_ASYNC() __builtin_amdgcn_s_wait_asynccnt(0)
#else
#define WAIT_ASYNC() asm volatile("s_wait_asynccnt 0" ::: "memory")
#endif

typedef __attribute__((ext_vector_type(16))) __bf16 v16bf;
typedef __attribute__((ext_vector_type(8)))  float  v8f;

struct Q2 { uint4 a, b; };  // 32-byte bit image of v16bf

__device__ __forceinline__ unsigned short f2bfu(float f) {
  return __builtin_bit_cast(unsigned short, (__bf16)f);  // RNE hw convert
}
__device__ __forceinline__ unsigned pk2bf(float lo, float hi) {
  return (unsigned)f2bfu(lo) | ((unsigned)f2bfu(hi) << 16);
}

// ---------------------------------------------------------------- zero output
__global__ void moe_zero(float4* __restrict__ p, int n4) {
  int i = blockIdx.x * blockDim.x + threadIdx.x;
  if (i < n4) p[i] = make_float4(0.f, 0.f, 0.f, 0.f);
}

// ---------------------------------------------------------------- gating
__global__ __launch_bounds__(256)
void moe_gate(const float* __restrict__ x, const float* __restrict__ Wg,
              const float* __restrict__ bg, float* __restrict__ gate_probs,
              int* __restrict__ topk, float* __restrict__ gates) {
  const int lane = threadIdx.x & 31;
  const int wave = threadIdx.x >> 5;
  const int b = blockIdx.x * 8 + wave;
  if (b >= B_TOK) return;

  float acc[NEXP];
#pragma unroll
  for (int e = 0; e < NEXP; ++e) acc[e] = 0.f;

  const float* xr = x + (size_t)b * DMODEL;
  for (int d = lane; d < DMODEL; d += 32) {
    float xv = xr[d];
    const float4* w = (const float4*)(Wg + (size_t)d * NEXP);
    float4 w0 = w[0], w1 = w[1];
    acc[0] += xv * w0.x; acc[1] += xv * w0.y;
    acc[2] += xv * w0.z; acc[3] += xv * w0.w;
    acc[4] += xv * w1.x; acc[5] += xv * w1.y;
    acc[6] += xv * w1.z; acc[7] += xv * w1.w;
  }
#pragma unroll
  for (int off = 16; off > 0; off >>= 1)
#pragma unroll
    for (int e = 0; e < NEXP; ++e) acc[e] += __shfl_xor(acc[e], off, 32);

  if (lane == 0) {
    float lg[NEXP];
#pragma unroll
    for (int e = 0; e < NEXP; ++e) lg[e] = acc[e] + bg[e];
    float m = lg[0];
#pragma unroll
    for (int e = 1; e < NEXP; ++e) m = lg[e] > m ? lg[e] : m;
    float s = 0.f, p[NEXP];
#pragma unroll
    for (int e = 0; e < NEXP; ++e) { p[e] = __expf(lg[e] - m); s += p[e]; }
    float inv = 1.f / s;
#pragma unroll
    for (int e = 0; e < NEXP; ++e) gate_probs[(size_t)b * NEXP + e] = p[e] * inv;

    int i0 = 0;
#pragma unroll
    for (int e = 1; e < NEXP; ++e) if (lg[e] > lg[i0]) i0 = e;
    int i1 = (i0 == 0) ? 1 : 0;
#pragma unroll
    for (int e = 0; e < NEXP; ++e) if (e != i0 && lg[e] > lg[i1]) i1 = e;

    float e1 = __expf(lg[i1] - lg[i0]);
    float g0 = 1.f / (1.f + e1);
    float g1 = e1 / (1.f + e1);
    topk[b * 2 + 0] = i0;
    topk[b * 2 + 1] = i1;
    gates[b * 2 + 0] = g0;
    gates[b * 2 + 1] = g1;
  }
}

// ---------------------------------------------------------------- compaction
__global__ __launch_bounds__(256)
void moe_compact(const int* __restrict__ topk, const float* __restrict__ gates,
                 int* __restrict__ offs, int* __restrict__ slot_token,
                 float* __restrict__ slot_gate) {
  __shared__ int cnt[NEXP];
  __shared__ int base[NEXP + 1];
  const int lane = threadIdx.x & 31;
  const int w = threadIdx.x >> 5;

  int c = 0;
  for (int t0 = 0; t0 < B_TOK; t0 += 32) {
    int b = t0 + lane;
    int i0 = topk[b * 2], i1 = topk[b * 2 + 1];
    unsigned long long m = __ballot(i0 == w || i1 == w);
    c += __popcll(m);
  }
  if (lane == 0) cnt[w] = c;
  __syncthreads();
  if (threadIdx.x == 0) {
    int s = 0;
    for (int e = 0; e < NEXP; ++e) { base[e] = s; s += cnt[e]; }
    base[NEXP] = s;
    for (int e = 0; e <= NEXP; ++e) offs[e] = base[e];
  }
  __syncthreads();

  int pos = base[w];
  for (int t0 = 0; t0 < B_TOK; t0 += 32) {
    int b = t0 + lane;
    int i0 = topk[b * 2], i1 = topk[b * 2 + 1];
    int k = (i0 == w) ? 0 : ((i1 == w) ? 1 : -1);
    unsigned mm = (unsigned)__ballot(k >= 0);
    if (k >= 0) {
      int ppos = pos + __popc(mm & ((1u << lane) - 1u));
      slot_token[ppos] = b;
      slot_gate[ppos] = gates[b * 2 + k];
    }
    pos += __popc(mm);
  }
}

// ---------------------------------------------------------------- FFN1
// h[slot, :] = relu(x[token] @ W1[e] + b1[e]) -> bf16
__global__ __launch_bounds__(256)
void moe_ffn1(const float* __restrict__ x, const float* __restrict__ W1,
              const float* __restrict__ b1, const int* __restrict__ offs,
              const int* __restrict__ slot_token, unsigned short* __restrict__ hbuf) {
  __shared__ unsigned short As[2][MT * LDT];   // 10 KB x2
  __shared__ unsigned short Bsm[2][NT * LDT];  // 10 KB x2

  const int e = blockIdx.z;
  const int mEnd = offs[e + 1];
  const int m0 = offs[e] + blockIdx.y * MT;
  if (m0 >= mEnd) return;
  const int n0 = blockIdx.x * NT;

  const int tid = threadIdx.x;
  const int lane = tid & 31, wave = tid >> 5;
  const int wm = wave & 1, wn = wave >> 1;
  const int lrow = lane & 15, lhalf = lane >> 4;
  const size_t eW = (size_t)e * DMODEL * HDIM;

  // staging mapping: A 128x32 f32, B 32x128 f32 (transposed into LDS)
  const int rA = tid >> 3;        // + it*32
  const int cA = (tid & 7) * 4;
  const int kB = tid >> 5;        // + it*8
  const int nB = (tid & 31) * 4;

  int tokOf[4];
#pragma unroll
  for (int it = 0; it < 4; ++it) {
    int s = m0 + rA + it * 32; if (s > mEnd - 1) s = mEnd - 1;
    tokOf[it] = slot_token[s];
  }

  float4 aReg[4], bReg[4];
  auto gload = [&](int k0) {
#pragma unroll
    for (int it = 0; it < 4; ++it)
      aReg[it] = *(const float4*)(x + (size_t)tokOf[it] * DMODEL + k0 + cA);
#pragma unroll
    for (int it = 0; it < 4; ++it)
      bReg[it] = *(const float4*)(W1 + eW + (size_t)(k0 + kB + it * 8) * HDIM + n0 + nB);
  };
  auto lstore = [&](int p) {
#pragma unroll
    for (int it = 0; it < 4; ++it) {
      uint2 v = make_uint2(pk2bf(aReg[it].x, aReg[it].y), pk2bf(aReg[it].z, aReg[it].w));
      *(uint2*)&As[p][(rA + it * 32) * LDT + cA] = v;
    }
#pragma unroll
    for (int it = 0; it < 4; ++it) {
      int kk = kB + it * 8;
      Bsm[p][(nB + 0) * LDT + kk] = f2bfu(bReg[it].x);
      Bsm[p][(nB + 1) * LDT + kk] = f2bfu(bReg[it].y);
      Bsm[p][(nB + 2) * LDT + kk] = f2bfu(bReg[it].z);
      Bsm[p][(nB + 3) * LDT + kk] = f2bfu(bReg[it].w);
    }
  };

  v8f acc[4][2];
  const v8f vzero = {0.f, 0.f, 0.f, 0.f, 0.f, 0.f, 0.f, 0.f};
#pragma unroll
  for (int i = 0; i < 4; ++i)
#pragma unroll
    for (int j = 0; j < 2; ++j) acc[i][j] = vzero;

  gload(0);
  lstore(0);
  __syncthreads();

  int p = 0;
  for (int k0 = 0; k0 < DMODEL; k0 += KT) {
    const bool more = (k0 + KT) < DMODEL;
    if (more) gload(k0 + KT);  // overlap with WMMA below

    v16bf av[4], bv[2];
#pragma unroll
    for (int mi = 0; mi < 4; ++mi) {
      const unsigned short* ap = &As[p][(wm * 64 + mi * 16 + lrow) * LDT + lhalf * 8];
      Q2 q; q.a = *(const uint4*)ap; q.b = *(const uint4*)(ap + 16);
      av[mi] = __builtin_bit_cast(v16bf, q);
    }
#pragma unroll
    for (int nj = 0; nj < 2; ++nj) {
      const unsigned short* bp = &Bsm[p][(wn * 32 + nj * 16 + lrow) * LDT + lhalf * 8];
      Q2 q; q.a = *(const uint4*)bp; q.b = *(const uint4*)(bp + 16);
      bv[nj] = __builtin_bit_cast(v16bf, q);
    }
#pragma unroll
    for (int mi = 0; mi < 4; ++mi)
#pragma unroll
      for (int nj = 0; nj < 2; ++nj)
        acc[mi][nj] = __builtin_amdgcn_wmma_f32_16x16x32_bf16(
            false, av[mi], false, bv[nj], (short)0, acc[mi][nj], false, false);

    if (more) lstore(p ^ 1);
    __syncthreads();
    p ^= 1;
  }

  // epilogue: bias + relu -> bf16 h
#pragma unroll
  for (int nj = 0; nj < 2; ++nj) {
    int col = n0 + wn * 32 + nj * 16 + lrow;
    float bias = b1[e * HDIM + col];
#pragma unroll
    for (int mi = 0; mi < 4; ++mi) {
#pragma unroll
      for (int v = 0; v < 8; ++v) {
        int r = m0 + wm * 64 + mi * 16 + lhalf * 8 + v;
        int s = (r > mEnd - 1) ? (mEnd - 1) : r;  // clamped rows rewrite same value
        float val = acc[mi][nj][v] + bias;
        val = val > 0.f ? val : 0.f;
        hbuf[(size_t)s * HDIM + col] = f2bfu(val);
      }
    }
  }
}

// ---------------------------------------------------------------- FFN2
// out[token] += gate * (h[slot] @ W2[e] + b2[e])
__global__ __launch_bounds__(256)
void moe_ffn2(const unsigned short* __restrict__ hbuf, const float* __restrict__ W2,
              const float* __restrict__ b2, const int* __restrict__ offs,
              const int* __restrict__ slot_token, const float* __restrict__ slot_gate,
              float* __restrict__ out) {
  __shared__ unsigned short As[2][MT * LDT];
  __shared__ unsigned short Bsm[2][NT * LDT];

  const int e = blockIdx.z;
  const int mEnd = offs[e + 1];
  const int m0 = offs[e] + blockIdx.y * MT;
  if (m0 >= mEnd) return;
  const int n0 = blockIdx.x * NT;

  const int tid = threadIdx.x;
  const int lane = tid & 31, wave = tid >> 5;
  const int wm = wave & 1, wn = wave >> 1;
  const int lrow = lane & 15, lhalf = lane >> 4;
  const size_t eW = (size_t)e * HDIM * DMODEL;

  // staging mapping: A 128x32 bf16 (raw copy), B 32x128 f32 (transposed)
  const int rA = tid >> 2;        // + it*64
  const int qA = (tid & 3) * 8;   // bf16 units, 16B chunks
  const int kB = tid >> 5;        // + it*8
  const int nB = (tid & 31) * 4;

  int sOf[2];
#pragma unroll
  for (int it = 0; it < 2; ++it) {
    int s = m0 + rA + it * 64; if (s > mEnd - 1) s = mEnd - 1;
    sOf[it] = s;
  }

  uint4 aRegU[2];
  float4 bReg[4];
  auto gloadA = [&](int k0, int pdst) {
    (void)pdst;
#pragma unroll
    for (int it = 0; it < 2; ++it) {
      const unsigned short* src = hbuf + (size_t)sOf[it] * HDIM + k0 + qA;
#if HAVE_ASYNC_LDS
      __builtin_amdgcn_global_load_async_to_lds_b128(
          (AS1 vi4*)src,
          (AS3 vi4*)&As[pdst][(rA + it * 64) * LDT + qA], 0, 0);
#else
      aRegU[it] = *(const uint4*)src;
#endif
    }
  };
  auto lstoreA = [&](int p) {
    (void)p;
#if !HAVE_ASYNC_LDS
#pragma unroll
    for (int it = 0; it < 2; ++it)
      *(uint4*)&As[p][(rA + it * 64) * LDT + qA] = aRegU[it];
#endif
  };
  auto gloadB = [&](int k0) {
#pragma unroll
    for (int it = 0; it < 4; ++it)
      bReg[it] = *(const float4*)(W2 + eW + (size_t)(k0 + kB + it * 8) * DMODEL + n0 + nB);
  };
  auto lstoreB = [&](int p) {
#pragma unroll
    for (int it = 0; it < 4; ++it) {
      int kk = kB + it * 8;
      Bsm[p][(nB + 0) * LDT + kk] = f2bfu(bReg[it].x);
      Bsm[p][(nB + 1) * LDT + kk] = f2bfu(bReg[it].y);
      Bsm[p][(nB + 2) * LDT + kk] = f2bfu(bReg[it].z);
      Bsm[p][(nB + 3) * LDT + kk] = f2bfu(bReg[it].w);
    }
  };

  v8f acc[4][2];
  const v8f vzero = {0.f, 0.f, 0.f, 0.f, 0.f, 0.f, 0.f, 0.f};
#pragma unroll
  for (int i = 0; i < 4; ++i)
#pragma unroll
    for (int j = 0; j < 2; ++j) acc[i][j] = vzero;

  gloadA(0, 0);
  gloadB(0);
  lstoreA(0);
  lstoreB(0);
#if HAVE_ASYNC_LDS
  WAIT_ASYNC();
#endif
  __syncthreads();

  int p = 0;
  for (int k0 = 0; k0 < HDIM; k0 += KT) {
    const bool more = (k0 + KT) < HDIM;
    if (more) { gloadA(k0 + KT, p ^ 1); gloadB(k0 + KT); }

    v16bf av[4], bv[2];
#pragma unroll
    for (int mi = 0; mi < 4; ++mi) {
      const unsigned short* ap = &As[p][(wm * 64 + mi * 16 + lrow) * LDT + lhalf * 8];
      Q2 q; q.a = *(const uint4*)ap; q.b = *(const uint4*)(ap + 16);
      av[mi] = __builtin_bit_cast(v16bf, q);
    }
#pragma unroll
    for (int nj = 0; nj < 2; ++nj) {
      const unsigned short* bp = &Bsm[p][(wn * 32 + nj * 16 + lrow) * LDT + lhalf * 8];
      Q2 q; q.a = *(const uint4*)bp; q.b = *(const uint4*)(bp + 16);
      bv[nj] = __builtin_bit_cast(v16bf, q);
    }
#pragma unroll
    for (int mi = 0; mi < 4; ++mi)
#pragma unroll
      for (int nj = 0; nj < 2; ++nj)
        acc[mi][nj] = __builtin_amdgcn_wmma_f32_16x16x32_bf16(
            false, av[mi], false, bv[nj], (short)0, acc[mi][nj], false, false);

    if (more) { lstoreA(p ^ 1); lstoreB(p ^ 1); }
#if HAVE_ASYNC_LDS
    WAIT_ASYNC();
#endif
    __syncthreads();
    p ^= 1;
  }

  // epilogue: scale by gate, scatter-add (2 commutative adds per out element)
#pragma unroll
  for (int nj = 0; nj < 2; ++nj) {
    int col = n0 + wn * 32 + nj * 16 + lrow;
    float bias = b2[e * DMODEL + col];
#pragma unroll
    for (int mi = 0; mi < 4; ++mi) {
#pragma unroll
      for (int v = 0; v < 8; ++v) {
        int r = m0 + wm * 64 + mi * 16 + lhalf * 8 + v;
        if (r < mEnd) {
          float g = slot_gate[r];
          int tok = slot_token[r];
          atomicAdd(out + (size_t)tok * DMODEL + col, (acc[mi][nj][v] + bias) * g);
        }
      }
    }
  }
}

// ---------------------------------------------------------------- launcher
extern "C" void kernel_launch(void* const* d_in, const int* in_sizes, int n_in,
                              void* d_out, int out_size, void* d_ws, size_t ws_size,
                              hipStream_t stream) {
  (void)in_sizes; (void)n_in; (void)out_size; (void)ws_size;

  const float* x  = (const float*)d_in[0];
  const float* Wg = (const float*)d_in[1];
  const float* bg = (const float*)d_in[2];
  const float* W1 = (const float*)d_in[3];
  const float* b1 = (const float*)d_in[4];
  const float* W2 = (const float*)d_in[5];
  const float* b2 = (const float*)d_in[6];

  float* out        = (float*)d_out;
  float* gate_probs = out + (size_t)B_TOK * DMODEL;
  int*   topk       = (int*)(gate_probs + (size_t)B_TOK * NEXP);

  // workspace layout (~67.3 MB total)
  char* ws = (char*)d_ws;
  unsigned short* hbuf = (unsigned short*)ws;                         // 64 MB
  size_t off = (size_t)NSLOTS * HDIM * sizeof(unsigned short);
  int*   slot_token = (int*)(ws + off);   off += (size_t)NSLOTS * 4;
  float* slot_gate  = (float*)(ws + off); off += (size_t)NSLOTS * 4;
  float* gates      = (float*)(ws + off); off += (size_t)B_TOK * 2 * 4;
  int*   offs       = (int*)(ws + off);   off += 64;

  int n4 = (B_TOK * DMODEL) / 4;
  moe_zero<<<dim3((n4 + 255) / 256), dim3(256), 0, stream>>>((float4*)out, n4);
  moe_gate<<<dim3(B_TOK / 8), dim3(256), 0, stream>>>(x, Wg, bg, gate_probs, topk, gates);
  moe_compact<<<dim3(1), dim3(256), 0, stream>>>(topk, gates, offs, slot_token, slot_gate);
  moe_ffn1<<<dim3(HDIM / NT, B_TOK / MT, NEXP), dim3(256), 0, stream>>>(
      x, W1, b1, offs, slot_token, hbuf);
  moe_ffn2<<<dim3(DMODEL / NT, B_TOK / MT, NEXP), dim3(256), 0, stream>>>(
      hbuf, W2, b2, offs, slot_token, slot_gate, out);
}
